// GATE_88940182766174
// MI455X (gfx1250) — matrix operand order repaired and checked by reading, not compiled, not run on hardware
//
#include <hip/hip_runtime.h>
#include <cstddef>

// ---------------- problem constants (match reference) ----------------
constexpr int NN  = 8192;    // nodes
constexpr int DIN = 561;     // input feature dim
constexpr int DH  = 256;     // hidden dim
constexpr int EDG = 262144;  // edges per view
constexpr int NCLS = 6;
constexpr int DZ  = 512;

// ---------------- types ----------------
typedef __attribute__((ext_vector_type(16))) __bf16          v16bf;
typedef __attribute__((ext_vector_type(8)))  float           v8f;
typedef __attribute__((ext_vector_type(4)))  float           f32x4;
typedef __attribute__((ext_vector_type(8)))  unsigned short  u16x8;

__device__ __forceinline__ unsigned short f32_to_bf16(float f) {
  union { float f; unsigned u; } v; v.f = f;
  unsigned u = v.u;
  u += 0x7FFFu + ((u >> 16) & 1u);           // round to nearest even
  return (unsigned short)(u >> 16);
}

__device__ __forceinline__ float wave_reduce_sum(float v) {
#pragma unroll
  for (int off = 16; off; off >>= 1) v += __shfl_xor(v, off, 32);
  return v;
}

__device__ __forceinline__ float block_reduce_sum(float v) {
  __shared__ float red[8];
  int lane = threadIdx.x & 31, w = threadIdx.x >> 5;
  v = wave_reduce_sum(v);
  if (!lane) red[w] = v;
  __syncthreads();
  int nw = blockDim.x >> 5;
  v = (threadIdx.x < (unsigned)nw) ? red[threadIdx.x] : 0.f;
  if (!w) v = wave_reduce_sum(v);
  __syncthreads();
  return v;   // valid in thread 0
}

// ---------------- WMMA GEMM: C[M,Nc] (+)= A[M,K] * op(B) (+bias) ----------
// BTRANS: B stored [Nc,K] (op(B)=B^T), else B[K,Nc].
// VEC:    staging uses 16B global_load_b128 (caller guarantees 16B-aligned
//         base pointers and ld%4==0).
// ACC:    split-K accumulation: blockIdx.z handles K-chunk [z*kSplit, ...),
//         stores via global_atomic_add_f32 into pre-zeroed C (bias ignored).
// 128 threads / block, 64x64 tile, 4 waves x (2x2) v_wmma_f32_16x16x32_bf16.
// LDS is pre-swizzled to fragment order: fragment = one 32B read per lane.
// Interior tiles: branch-free staging + packed ds_store_b64; edge tiles take
// a guarded scalar path behind a uniform branch.
template <bool BTRANS, bool VEC, bool ACC>
__global__ __launch_bounds__(128)
void wmma_gemm_bf16(const float* __restrict__ A, const float* __restrict__ B,
                    float* __restrict__ C, const float* __restrict__ bias,
                    int M, int K, int Nc, int lda, int ldb, int ldc, int kSplit)
{
  __shared__ __attribute__((aligned(16))) unsigned short As[4 * 32 * 16]; // 4KB
  __shared__ __attribute__((aligned(16))) unsigned short Bs[4 * 32 * 16]; // 4KB

  const int tid  = threadIdx.x;
  const int lane = tid & 31;
  const int wave = tid >> 5;
  const int l  = lane & 15;
  const int hi = lane >> 4;
  const int bm = blockIdx.y * 64;
  const int bn = blockIdx.x * 64;
  const int wm = (wave >> 1) * 32;   // 0 or 32
  const int wn = (wave & 1) * 32;    // 0 or 32

  // fast-path mapping: 4 consecutive K per thread (4 consecutive j in LDS)
  const int kA   = (tid & 7) * 4;                  // 0,4,...,28
  const int rA   = tid >> 3;                       // 0..15
  const int jA   = kA & 15;                        // k & 15, 4 consecutive
  const int khiA = kA >> 4;
  // fast-path mapping for normal-B: 4 consecutive n per thread
  const int nB = (tid & 15) * 4;                   // 0..60
  const int kB = tid >> 4;                         // 0..7

  v8f acc[2][2];
  acc[0][0] = (v8f){}; acc[0][1] = (v8f){};
  acc[1][0] = (v8f){}; acc[1][1] = (v8f){};

  const bool fullM = (bm + 64 <= M);
  const bool fullN = (bn + 64 <= Nc);

  const int kBegin = blockIdx.z * kSplit;
  const int kEnd   = (kBegin + kSplit < K) ? (kBegin + kSplit) : K;

  for (int k0 = kBegin; k0 < kEnd; k0 += 32) {
    const bool fullK = (k0 + 32 <= K);

    // ---------------- stage A tile 64x32 ----------------
    if (fullK && fullM) {
#pragma unroll
      for (int i = 0; i < 4; ++i) {
        const int r = rA + i * 16;                 // 0..63
        const float* src = A + (size_t)(bm + r) * lda + (k0 + kA);
        float t0, t1, t2, t3;
        if (VEC) { const f32x4 v4 = *(const f32x4*)src; t0 = v4[0]; t1 = v4[1]; t2 = v4[2]; t3 = v4[3]; }
        else     { t0 = src[0]; t1 = src[1]; t2 = src[2]; t3 = src[3]; }
        unsigned long long pk =  (unsigned long long)f32_to_bf16(t0)
                              | ((unsigned long long)f32_to_bf16(t1) << 16)
                              | ((unsigned long long)f32_to_bf16(t2) << 32)
                              | ((unsigned long long)f32_to_bf16(t3) << 48);
        *(unsigned long long*)&As[(((r >> 4) * 32 + khiA * 16 + (r & 15)) << 4) + jA] = pk;
      }
    } else {
      const int k = tid & 31;
      const int khi = k >> 4;
      const int j = k & 15;
#pragma unroll
      for (int i = 0; i < 16; ++i) {
        const int r = (tid >> 5) + i * 4;          // 0..63
        const int gr = bm + r, gk = k0 + k;
        const float v = (gr < M && gk < K) ? A[(size_t)gr * lda + gk] : 0.f;
        As[(((r >> 4) * 32 + khi * 16 + (r & 15)) << 4) + j] = f32_to_bf16(v);
      }
    }

    // ---------------- stage B tile 32x64 ----------------
    if (BTRANS) {
      if (fullK && fullN) {
#pragma unroll
        for (int i = 0; i < 4; ++i) {
          const int n = rA + i * 16;               // 0..63
          const float* src = B + (size_t)(bn + n) * ldb + (k0 + kA);
          float t0, t1, t2, t3;
          if (VEC) { const f32x4 v4 = *(const f32x4*)src; t0 = v4[0]; t1 = v4[1]; t2 = v4[2]; t3 = v4[3]; }
          else     { t0 = src[0]; t1 = src[1]; t2 = src[2]; t3 = src[3]; }
          unsigned long long pk =  (unsigned long long)f32_to_bf16(t0)
                                | ((unsigned long long)f32_to_bf16(t1) << 16)
                                | ((unsigned long long)f32_to_bf16(t2) << 32)
                                | ((unsigned long long)f32_to_bf16(t3) << 48);
          *(unsigned long long*)&Bs[(((n >> 4) * 32 + khiA * 16 + (n & 15)) << 4) + jA] = pk;
        }
      } else {
        const int n = tid & 63, kb = tid >> 6;
        const int fn = n >> 4, lcol = n & 15;
#pragma unroll
        for (int i = 0; i < 16; ++i) {
          const int kk = kb + i * 2;               // 0..31
          const int gk = k0 + kk, gn = bn + n;
          const float v = (gk < K && gn < Nc) ? B[(size_t)gn * ldb + gk] : 0.f;
          Bs[(((fn * 32) + (kk >> 4) * 16 + lcol) << 4) + (kk & 15)] = f32_to_bf16(v);
        }
      }
    } else {
      if (fullK && fullN) {
        const int fn = nB >> 4, lc = nB & 15;      // nB..nB+3 share fn
#pragma unroll
        for (int i = 0; i < 4; ++i) {
          const int k = kB + i * 8;                // 0..31
          const int khi = k >> 4, j = k & 15;
          const float* src = B + (size_t)(k0 + k) * ldb + (bn + nB);
          float t[4];
          if (VEC) { const f32x4 v4 = *(const f32x4*)src; t[0] = v4[0]; t[1] = v4[1]; t[2] = v4[2]; t[3] = v4[3]; }
          else     { t[0] = src[0]; t[1] = src[1]; t[2] = src[2]; t[3] = src[3]; }
#pragma unroll
          for (int q = 0; q < 4; ++q)
            Bs[(((fn * 32) + khi * 16 + (lc + q)) << 4) + j] = f32_to_bf16(t[q]);
        }
      } else {
        const int n = tid & 63, kb = tid >> 6;
        const int fn = n >> 4, lcol = n & 15;
#pragma unroll
        for (int i = 0; i < 16; ++i) {
          const int kk = kb + i * 2;
          const int gk = k0 + kk, gn = bn + n;
          const float v = (gk < K && gn < Nc) ? B[(size_t)gk * ldb + gn] : 0.f;
          Bs[(((fn * 32) + (kk >> 4) * 16 + lcol) << 4) + (kk & 15)] = f32_to_bf16(v);
        }
      }
    }
    __syncthreads();

    // ---- fragment loads (contiguous 32B per lane) + 4 WMMAs ----
    const int frA = wm >> 4;     // 0 or 2
    const int fnB = wn >> 4;     // 0 or 2
    v16bf afr[2], bfr[2];
    afr[0] = *(const v16bf*)&As[((frA * 32) + lane) << 4];
    afr[1] = *(const v16bf*)&As[(((frA + 1) * 32) + lane) << 4];
    bfr[0] = *(const v16bf*)&Bs[((fnB * 32) + lane) << 4];
    bfr[1] = *(const v16bf*)&Bs[(((fnB + 1) * 32) + lane) << 4];

#pragma unroll
    for (int mi = 0; mi < 2; ++mi)
#pragma unroll
      for (int ni = 0; ni < 2; ++ni)
        acc[mi][ni] = __builtin_amdgcn_wmma_f32_16x16x32_bf16(
            false, afr[mi], false, bfr[ni], (short)0, acc[mi][ni], false, false);

    __syncthreads();
  }

  // ---- store C: plain (+bias) or split-K atomic accumulate ----
#pragma unroll
  for (int mi = 0; mi < 2; ++mi)
#pragma unroll
    for (int ni = 0; ni < 2; ++ni) {
      const int gn = bn + wn + ni * 16 + l;
      if (gn >= Nc) continue;
      const float bv = (!ACC && bias) ? bias[gn] : 0.f;
      const int gmb = bm + wm + mi * 16 + 8 * hi;
#pragma unroll
      for (int r = 0; r < 8; ++r) {
        const int gm = gmb + r;
        if (gm < M) {
          if (ACC) atomicAdd(&C[(size_t)gm * ldc + gn], acc[mi][ni][r]);
          else     C[(size_t)gm * ldc + gn] = acc[mi][ni][r] + bv;
        }
      }
    }
}

// ---------------- WMMA GEMM with pre-converted bf16 A ----------------------
// A is bf16 [M,K] row-major; B f32 [K,Nc]. Full tiles only: caller guarantees
// M%64==0, K%32==0 (and kSplit%32==0), Nc%64==0, 16B-aligned pointers.
// A staging is pure data movement: 2x global b128 + 2x ds_store_b128 per
// thread per tile (8 consecutive bf16 land on 16 contiguous LDS bytes of the
// fragment-swizzled layout). bf16 A (134MB) fits in MI455X's 192MB L2, so
// all re-reads (N-blocks, split-K, second view) are L2 hits.
template <bool ACC>
__global__ __launch_bounds__(128)
void wmma_gemm_bf16A(const unsigned short* __restrict__ A, const float* __restrict__ B,
                     float* __restrict__ C, int M, int K, int Nc,
                     int lda, int ldb, int ldc, int kSplit)
{
  __shared__ __attribute__((aligned(16))) unsigned short As[4 * 32 * 16];
  __shared__ __attribute__((aligned(16))) unsigned short Bs[4 * 32 * 16];

  const int tid  = threadIdx.x;
  const int lane = tid & 31;
  const int wave = tid >> 5;
  const int l  = lane & 15;
  const int hi = lane >> 4;
  const int bm = blockIdx.y * 64;
  const int bn = blockIdx.x * 64;
  const int wm = (wave >> 1) * 32;
  const int wn = (wave & 1) * 32;

  // A staging: 8 consecutive bf16 (16B) per load
  const int k8   = (tid & 3) * 8;                 // 0,8,16,24
  const int rA   = tid >> 2;                      // 0..31
  const int jb   = k8 & 15;                       // 0 or 8
  const int khi8 = k8 >> 4;
  // B staging (normal, vectorized)
  const int nB = (tid & 15) * 4;
  const int kB = tid >> 4;

  v8f acc[2][2];
  acc[0][0] = (v8f){}; acc[0][1] = (v8f){};
  acc[1][0] = (v8f){}; acc[1][1] = (v8f){};

  const int kBegin = blockIdx.z * kSplit;
  const int kEnd   = (kBegin + kSplit < K) ? (kBegin + kSplit) : K;

  for (int k0 = kBegin; k0 < kEnd; k0 += 32) {
    // ---- A: bf16 copy, no conversion ----
#pragma unroll
    for (int it = 0; it < 2; ++it) {
      const int r = rA + it * 32;                 // 0..63
      const u16x8 v = *(const u16x8*)(A + (size_t)(bm + r) * lda + (k0 + k8));
      *(u16x8*)&As[(((r >> 4) * 32 + khi8 * 16 + (r & 15)) << 4) + jb] = v;
    }
    // ---- B: f32 -> bf16 ----
    {
      const int fn = nB >> 4, lc = nB & 15;
#pragma unroll
      for (int i = 0; i < 4; ++i) {
        const int k = kB + i * 8;                 // 0..31
        const int khi = k >> 4, j = k & 15;
        const f32x4 v4 = *(const f32x4*)(B + (size_t)(k0 + k) * ldb + (bn + nB));
#pragma unroll
        for (int q = 0; q < 4; ++q)
          Bs[(((fn * 32) + khi * 16 + (lc + q)) << 4) + j] = f32_to_bf16(v4[q]);
      }
    }
    __syncthreads();

    const int frA = wm >> 4;
    const int fnB = wn >> 4;
    v16bf afr[2], bfr[2];
    afr[0] = *(const v16bf*)&As[((frA * 32) + lane) << 4];
    afr[1] = *(const v16bf*)&As[(((frA + 1) * 32) + lane) << 4];
    bfr[0] = *(const v16bf*)&Bs[((fnB * 32) + lane) << 4];
    bfr[1] = *(const v16bf*)&Bs[(((fnB + 1) * 32) + lane) << 4];

#pragma unroll
    for (int mi = 0; mi < 2; ++mi)
#pragma unroll
      for (int ni = 0; ni < 2; ++ni)
        acc[mi][ni] = __builtin_amdgcn_wmma_f32_16x16x32_bf16(
            false, afr[mi], false, bfr[ni], (short)0, acc[mi][ni], false, false);

    __syncthreads();
  }

#pragma unroll
  for (int mi = 0; mi < 2; ++mi)
#pragma unroll
    for (int ni = 0; ni < 2; ++ni) {
      const int gn = bn + wn + ni * 16 + l;
      const int gmb = bm + wm + mi * 16 + 8 * hi;
#pragma unroll
      for (int r = 0; r < 8; ++r) {
        const int gm = gmb + r;
        if (ACC) atomicAdd(&C[(size_t)gm * ldc + gn], acc[mi][ni][r]);
        else     C[(size_t)gm * ldc + gn] = acc[mi][ni][r];
      }
    }
}

// ---------------- bulk f32 -> bf16 (4 elements / thread) -------------------
__global__ __launch_bounds__(256)
void cvt_bf16_kernel(const float* __restrict__ src, unsigned short* __restrict__ dst,
                     long long n4)
{
  for (long long i = (long long)blockIdx.x * blockDim.x + threadIdx.x;
       i < n4; i += (long long)gridDim.x * blockDim.x) {
    const f32x4 v = *(const f32x4*)(src + i * 4);
    const unsigned long long pk =  (unsigned long long)f32_to_bf16(v[0])
                                | ((unsigned long long)f32_to_bf16(v[1]) << 16)
                                | ((unsigned long long)f32_to_bf16(v[2]) << 32)
                                | ((unsigned long long)f32_to_bf16(v[3]) << 48);
    *(unsigned long long*)(dst + i * 4) = pk;
  }
}

// ---------------- HC[i] -= weight[i][i] * Henc[i]  (coef diag correction) ---
__global__ __launch_bounds__(256)
void hc_diag_fixup_kernel(const float* __restrict__ weight,
                          const float* __restrict__ Henc, float* __restrict__ HC)
{
  const int i = blockIdx.x;
  const int c = threadIdx.x;                 // DH == 256
  const float wii = weight[(size_t)i * NN + i];
  HC[(size_t)i * DH + c] -= wii * Henc[(size_t)i * DH + c];
}

// ---------------- coef = weight - diag; S_Regular, Cq_loss -----------------
__global__ __launch_bounds__(256)
void coef_kernel(const float* __restrict__ weight, const float* __restrict__ Theta,
                 float* __restrict__ coef, float* accS, float* accCq)
{
  __shared__ float Ts[64 * 65];
  const int i0 = blockIdx.y * 64;
  const int j0 = blockIdx.x * 64;
  const int tid = threadIdx.x;

#pragma unroll
  for (int t = 0; t < 16; ++t) {
    const int idx = tid + t * 256;
    const int a = idx >> 6;             // local j
    const int b = idx & 63;             // local i
    Ts[a * 65 + b] = Theta[(size_t)(j0 + a) * NN + (i0 + b)];
  }
  __syncthreads();

  float sS = 0.f, sC = 0.f;
#pragma unroll
  for (int t = 0; t < 16; ++t) {
    const int idx = tid + t * 256;
    const int r  = idx >> 6;            // local i
    const int cj = idx & 63;            // local j
    const int gi = i0 + r, gj = j0 + cj;
    const size_t off = (size_t)gi * NN + gj;
    const float c = (gi == gj) ? 0.f : weight[off];
    coef[off] = c;
    sS += fabsf(c);
    sC += fabsf(c * Ts[cj * 65 + r]);   // Theta[gj][gi]
  }
  sS = block_reduce_sum(sS);
  sC = block_reduce_sum(sC);
  if (tid == 0) { atomicAdd(accS, sS); atomicAdd(accCq, sC); }
}

// ---------------- f1 = H @ v0, f2 = H @ v1 (one wave per row) --------------
__global__ __launch_bounds__(128)
void rowdot2_kernel(const float* __restrict__ H, const float* __restrict__ v0,
                    const float* __restrict__ v1, float* f1, float* f2)
{
  const int row  = blockIdx.x * 4 + (threadIdx.x >> 5);
  const int lane = threadIdx.x & 31;
  if (row >= NN) return;
  const float* h = H + (size_t)row * DH;
  float s1 = 0.f, s2 = 0.f;
  for (int k = lane; k < DH; k += 32) { const float hv = h[k]; s1 += hv * v0[k]; s2 += hv * v1[k]; }
  s1 = wave_reduce_sum(s1); s2 = wave_reduce_sum(s2);
  if (!lane) { f1[row] = s1; f2[row] = s2; }
}

// ---------------- edge attention: sigmoid logits + segment softmax ----------
__global__ __launch_bounds__(256)
void edge_logits_kernel(const int* __restrict__ row, const int* __restrict__ col,
                        const float* __restrict__ f1, const float* __restrict__ f2,
                        float* __restrict__ u, float* __restrict__ rmax)
{
  const int e = blockIdx.x * blockDim.x + threadIdx.x;
  if (e >= EDG) return;
  const float x = f1[row[e]] + f2[col[e]];
  const float s = 1.f / (1.f + __expf(-x));      // in (0,1): uint order == float order
  u[e] = s;
  atomicMax((unsigned*)&rmax[row[e]], __float_as_uint(s));
}

__global__ __launch_bounds__(256)
void edge_exp_kernel(const int* __restrict__ row, float* __restrict__ u,
                     const float* __restrict__ rmax, float* __restrict__ rsum)
{
  const int e = blockIdx.x * blockDim.x + threadIdx.x;
  if (e >= EDG) return;
  const float ex = __expf(u[e] - rmax[row[e]]);
  u[e] = ex;
  atomicAdd(&rsum[row[e]], ex);
}

__global__ __launch_bounds__(256)
void edge_norm_kernel(const int* __restrict__ row, float* __restrict__ u,
                      const float* __restrict__ rsum)
{
  const int e = blockIdx.x * blockDim.x + threadIdx.x;
  if (e >= EDG) return;
  u[e] = u[e] / (rsum[row[e]] + 1e-12f);
}

// ---------------- out[row] += att * src[col]  (one wave per edge) -----------
__global__ __launch_bounds__(256)
void spmm_scatter_kernel(const int* __restrict__ row, const int* __restrict__ col,
                         const float* __restrict__ att, const float* __restrict__ src,
                         float* __restrict__ out, int d)
{
  const int e = blockIdx.x * 8 + (threadIdx.x >> 5);
  const int lane = threadIdx.x & 31;
  if (e >= EDG) return;
  const float a = att[e];
  const float* s = src + (size_t)col[e] * d;
  float* o = out + (size_t)row[e] * d;
  for (int k = lane; k < d; k += 32)
    atomicAdd(&o[k], a * s[k]);
}

// ---------------- acc += scale * sum((a-b)^2) ------------------------------
__global__ __launch_bounds__(256)
void sqdiff_kernel(const float* __restrict__ a, const float* __restrict__ b,
                   long long n, float* acc, float scale)
{
  float s = 0.f;
  for (long long i = (long long)blockIdx.x * blockDim.x + threadIdx.x;
       i < n; i += (long long)gridDim.x * blockDim.x) {
    const float d = a[i] - b[i];
    s += d * d;
  }
  s = block_reduce_sum(s);
  if (threadIdx.x == 0) atomicAdd(acc, scale * s);
}

// ---------------- structure loss: sum softplus(-<Henc[S],Henc[R]>) ----------
__global__ __launch_bounds__(128)
void struct_loss_kernel(const int* __restrict__ S, const int* __restrict__ R,
                        const float* __restrict__ Henc, float* acc)
{
  const int e    = blockIdx.x * 4 + (threadIdx.x >> 5);
  const int lane = threadIdx.x & 31;
  if (e >= EDG) return;
  const float* hs = Henc + (size_t)S[e] * DH;
  const float* hr = Henc + (size_t)R[e] * DH;
  float s = 0.f;
  for (int k = lane; k < DH; k += 32) s += hs[k] * hr[k];
  s = wave_reduce_sum(s);
  if (!lane) {
    const float sp = fmaxf(-s, 0.f) + log1pf(__expf(-fabsf(s)));  // -log_sigmoid
    atomicAdd(acc, sp);
  }
}

// ---------------- dense CE: z = t1 @ zw^T + zb, sum -log_softmax[p] ---------
__global__ __launch_bounds__(128)
void ce_kernel(const float* __restrict__ t1, const float* __restrict__ zw,
               const float* __restrict__ zb, const int* __restrict__ p, float* acc)
{
  const int node = blockIdx.x * 4 + (threadIdx.x >> 5);
  const int lane = threadIdx.x & 31;
  if (node >= NN) return;
  const float* t = t1 + (size_t)node * DZ;
  float z[NCLS];
#pragma unroll
  for (int c = 0; c < NCLS; ++c) {
    const float* w = zw + (size_t)c * DZ;
    float s = 0.f;
    for (int k = lane; k < DZ; k += 32) s += t[k] * w[k];
    s = wave_reduce_sum(s);
    z[c] = s + zb[c];
  }
  if (!lane) {
    float m = z[0];
#pragma unroll
    for (int c = 1; c < NCLS; ++c) m = fmaxf(m, z[c]);
    float se = 0.f;
#pragma unroll
    for (int c = 0; c < NCLS; ++c) se += __expf(z[c] - m);
    const float lse = m + __logf(se);
    atomicAdd(acc, lse - z[p[node]]);
  }
}

// ---------------- combine scalars ------------------------------------------
__global__ void finalize_kernel(const float* __restrict__ acc, float* __restrict__ out)
{
  const float feat = acc[0], SE = acc[1], st = acc[2], cons = acc[3];
  const float sreg = acc[4], cq = acc[5], dense = acc[6];
  const long long nn = (long long)NN * NN;
  out[0] = feat + st + 10.f * SE + 0.01f * cons + sreg;                       // pre_loss
  out[1] = 0.01f * feat + st + 10.f * SE + 0.001f * cons + sreg
           + 5.f * cq + 5.f * dense;                                          // loss
  out[2] = dense; out[3] = feat; out[4] = st; out[5] = SE;
  out[6 + nn] = cons; out[7 + nn] = sreg; out[8 + nn] = cq;
}

// ===========================================================================
extern "C" void kernel_launch(void* const* d_in, const int* in_sizes, int n_in,
                              void* d_out, int out_size, void* d_ws, size_t ws_size,
                              hipStream_t stream)
{
  (void)in_sizes; (void)n_in; (void)out_size;

  // ---- inputs (setup_inputs order) ----
  const float* X      = (const float*)d_in[0];
  const float* X2     = (const float*)d_in[1];
  const float* Theta  = (const float*)d_in[2];
  const int*   A_idx  = (const int*)  d_in[3];
  const int*   A2_idx = (const int*)  d_in[5];
  const int*   S      = (const int*)  d_in[7];
  const int*   R      = (const int*)  d_in[8];
  const int*   S2     = (const int*)  d_in[9];
  const int*   R2     = (const int*)  d_in[10];
  const int*   p      = (const int*)  d_in[11];
  const float* W0     = (const float*)d_in[13];
  const float* v00    = (const float*)d_in[14];
  const float* v10    = (const float*)d_in[15];
  const float* weight = (const float*)d_in[16];
  const float* fc1_w  = (const float*)d_in[17];
  const float* fc1_b  = (const float*)d_in[18];
  const float* fcz_w  = (const float*)d_in[19];
  const float* fcz_b  = (const float*)d_in[20];
  const float* fc2_w  = (const float*)d_in[21];
  const float* fc2_b  = (const float*)d_in[22];
  const float* fcz2_w = (const float*)d_in[23];
  const float* fcz2_b = (const float*)d_in[24];

  // ---- output layout (tuple order, flat) ----
  float* out   = (float*)d_out;
  float* coef  = out + 6;                                   // N*N
  float* Henc1 = out + 9 + (size_t)NN * NN;                 // N*DH
  float* Henc2 = Henc1 + (size_t)NN * DH;                   // N*DH

  // ---- workspace layout (all GEMM operands 16B-aligned) ----
  float* ws    = (float*)d_ws;
  float* acc   = ws;                           // 64 floats reserved (7 used)
  float* H     = acc + 64;                     // N*DH
  float* HC    = H     + (size_t)NN * DH;      // N*DH
  float* HencW = HC    + (size_t)NN * DH;      // N*DH  (aligned H_enc)
  float* bufA  = HencW + (size_t)NN * DH;      // N*DIN (pre / t1)
  float* bufB  = bufA  + (size_t)NN * DIN;     // N*DIN (X_recon)
  float* f1    = bufB  + (size_t)NN * DIN;     // N
  float* f2    = f1 + NN;                      // N
  float* rmax  = f2 + NN;                      // N
  float* rsum  = rmax + NN;                    // N
  float* att   = rsum + NN;                    // E

  // optional bf16 copy of `weight` (fits MI455X 192MB L2), gated on ws_size
  const size_t baseFloats = 64 + 3 * (size_t)NN * DH + 2 * (size_t)NN * DIN
                            + 4 * (size_t)NN + (size_t)EDG;
  unsigned short* Wbf = (unsigned short*)(ws + baseFloats);
  const bool useBf16A = ws_size >= baseFloats * sizeof(float)
                                  + (size_t)NN * NN * sizeof(unsigned short);

  const dim3 blk128(128), blk256(256);
  auto ggrid = [](int m, int n) { return dim3((unsigned)((n + 63) / 64), (unsigned)((m + 63) / 64)); };

  hipMemsetAsync(acc, 0, 64 * sizeof(float), stream);

  // ---- coef / S_Regular / Cq_loss (coef output only; GEMMs use `weight`) ----
  coef_kernel<<<dim3(NN / 64, NN / 64), blk256, 0, stream>>>(weight, Theta, coef, &acc[4], &acc[5]);

  // ---- one-time weight -> bf16 conversion (shared by both views) ----
  if (useBf16A)
    cvt_bf16_kernel<<<8192, blk256, 0, stream>>>(weight, Wbf, (long long)NN * NN / 4);

  for (int view = 0; view < 2; ++view) {
    const float* Xv   = view == 0 ? X  : X2;
    const int*   idx  = view == 0 ? A_idx : A2_idx;
    const int*   rowp = idx;
    const int*   colp = idx + EDG;
    const int*   Sv   = view == 0 ? S  : S2;
    const int*   Rv   = view == 0 ? R  : R2;
    const float* fcw  = view == 0 ? fc1_w  : fc2_w;
    const float* fcb  = view == 0 ? fc1_b  : fc2_b;
    const float* zw   = view == 0 ? fcz_w  : fcz2_w;
    const float* zb   = view == 0 ? fcz_b  : fcz2_b;
    float*       HencOut = view == 0 ? Henc1 : Henc2;

    // H = Xv @ W0                    [N,DIN]x[DIN,DH]   (X rows unaligned -> scalar staging)
    wmma_gemm_bf16<false, false, false><<<ggrid(NN, DH), blk128, 0, stream>>>(
        Xv, W0, H, nullptr, NN, DIN, DH, DIN, DH, DH, DIN + 32);

    // attention
    rowdot2_kernel<<<NN / 4, blk128, 0, stream>>>(H, v00, v10, f1, f2);
    hipMemsetAsync(rmax, 0, NN * sizeof(float), stream);
    hipMemsetAsync(rsum, 0, NN * sizeof(float), stream);
    edge_logits_kernel<<<EDG / 256, blk256, 0, stream>>>(rowp, colp, f1, f2, att, rmax);
    edge_exp_kernel<<<EDG / 256, blk256, 0, stream>>>(rowp, att, rmax, rsum);
    edge_norm_kernel<<<EDG / 256, blk256, 0, stream>>>(rowp, att, rsum);

    // H_enc = segment_sum(att * H[col])  -> aligned ws buffer, then copy to output
    hipMemsetAsync(HencW, 0, (size_t)NN * DH * sizeof(float), stream);
    spmm_scatter_kernel<<<EDG / 8, blk256, 0, stream>>>(rowp, colp, att, H, HencW, DH);
    hipMemcpyAsync(HencOut, HencW, (size_t)NN * DH * sizeof(float),
                   hipMemcpyDeviceToDevice, stream);

    // HC = coef @ H_enc  ==  weight @ H_enc - diag(weight) .* H_enc
    // dominant GEMM: split-K x4 (2048 blocks), atomic accumulate
    hipMemsetAsync(HC, 0, (size_t)NN * DH * sizeof(float), stream);
    if (useBf16A)
      wmma_gemm_bf16A<true><<<dim3(DH / 64, NN / 64, 4), blk128, 0, stream>>>(
          Wbf, HencW, HC, NN, NN, DH, NN, DH, DH, NN / 4);
    else
      wmma_gemm_bf16<false, true, true><<<dim3(DH / 64, NN / 64, 4), blk128, 0, stream>>>(
          weight, HencW, HC, nullptr, NN, NN, DH, NN, DH, DH, NN / 4);
    hc_diag_fixup_kernel<<<NN, blk256, 0, stream>>>(weight, HencW, HC);

    // SE_loss += 0.5 * ||H_enc - HC||^2
    sqdiff_kernel<<<2048, blk256, 0, stream>>>(HencW, HC, (long long)NN * DH, &acc[1], 0.5f);

    // pre = HC @ W0^T                [N,DH]x[DH,DIN] (B stored [DIN,DH], aligned)
    wmma_gemm_bf16<true, true, false><<<ggrid(NN, DIN), blk128, 0, stream>>>(
        HC, W0, bufA, nullptr, NN, DH, DIN, DH, DH, DIN, DH + 32);

    // X_recon = segment_sum(att * pre[col])
    hipMemsetAsync(bufB, 0, (size_t)NN * DIN * sizeof(float), stream);
    spmm_scatter_kernel<<<EDG / 8, blk256, 0, stream>>>(rowp, colp, att, bufA, bufB, DIN);

    // features_loss += ||Xv - X_recon||^2
    sqdiff_kernel<<<2048, blk256, 0, stream>>>(Xv, bufB, (long long)NN * DIN, &acc[0], 1.0f);

    // structure_loss += sum -log_sigmoid(<Henc[S],Henc[R]>)
    struct_loss_kernel<<<EDG / 4, blk128, 0, stream>>>(Sv, Rv, HencW, &acc[2]);

    // t1 = Henc @ fcw^T + fcb        [N,DH]x[DH,DZ] (B stored [DZ,DH], aligned)
    wmma_gemm_bf16<true, true, false><<<ggrid(NN, DZ), blk128, 0, stream>>>(
        HencW, fcw, bufA, fcb, NN, DH, DZ, DH, DH, DZ, DH + 32);

    // dense_loss += CE(t1 @ zw^T + zb, p)
    ce_kernel<<<NN / 4, blk128, 0, stream>>>(bufA, zw, zb, p, &acc[6]);
  }

  // consistent_loss = ||Henc1 - Henc2||^2 (output copies)
  sqdiff_kernel<<<2048, blk256, 0, stream>>>(Henc1, Henc2, (long long)NN * DH, &acc[3], 1.0f);

  finalize_kernel<<<1, 1, 0, stream>>>(acc, out);
}
